// LSTM_Block_73048803770861
// MI455X (gfx1250) — matrix operand order repaired and checked by reading, not compiled, not run on hardware
//
#include <hip/hip_runtime.h>

#define DI  2048   // IN
#define DH  2048   // H
#define DB  8192   // batch
#define H4  8192   // 4*H
#define KD  2048   // per-matrix K depth (IN == H)
#define KT  4096   // concatenated K: [x | h] vs [Wi^T ; Wh^T]
#define NT  128    // number of 32-deep k tiles (KT/32)

typedef __attribute__((ext_vector_type(16))) __bf16 v16bf;
typedef __attribute__((ext_vector_type(8)))  float  v8f;

static __device__ __forceinline__ unsigned short f2bf(float f) {
  unsigned u = __float_as_uint(f);
  u += 0x7FFFu + ((u >> 16) & 1u);   // round to nearest even
  return (unsigned short)(u >> 16);
}

static __device__ __forceinline__ float fsigmoid(float x) {
  return __builtin_amdgcn_rcpf(1.f + __expf(-x));
}
static __device__ __forceinline__ float ftanh(float x) {
  return 2.f * __builtin_amdgcn_rcpf(1.f + __expf(-2.f * x)) - 1.f;
}

// ---------------- fp32 -> bf16 elementwise ----------------
__global__ __launch_bounds__(256)
void cvt_bf16(const float* __restrict__ in, unsigned short* __restrict__ out) {
  size_t i = ((size_t)blockIdx.x * blockDim.x + threadIdx.x) * 4;
  float4 v = *(const float4*)(in + i);
  uint2 p;
  p.x = (unsigned)f2bf(v.x) | ((unsigned)f2bf(v.y) << 16);
  p.y = (unsigned)f2bf(v.z) | ((unsigned)f2bf(v.w) << 16);
  *(uint2*)(out + i) = p;
}

// ---------------- fp32 [R][C] -> bf16 [C][R] (transpose) ----------------
__global__ __launch_bounds__(256)
void cvt_tr_bf16(const float* __restrict__ in, unsigned short* __restrict__ out,
                 int R, int C) {
  __shared__ unsigned short tile[32][33];
  int c0 = blockIdx.x * 32, r0 = blockIdx.y * 32;
  int tx = threadIdx.x, ty = threadIdx.y;           // tx: 0..31, ty: 0..7
  #pragma unroll
  for (int i = 0; i < 32; i += 8)
    tile[ty + i][tx] = f2bf(in[(size_t)(r0 + ty + i) * C + (c0 + tx)]);
  __syncthreads();
  #pragma unroll
  for (int i = 0; i < 32; i += 8)
    out[(size_t)(c0 + ty + i) * R + (r0 + tx)] = tile[tx][ty + i];
}

// async DMA: global (16B) -> LDS, tracked by ASYNCcnt (CDNA5)
static __device__ __forceinline__ void async_ld_b128(unsigned lds_byte_off,
                                                     const unsigned short* g) {
  asm volatile("global_load_async_to_lds_b128 %0, %1, off"
               :: "v"(lds_byte_off), "v"((unsigned long long)(uintptr_t)g)
               : "memory");
}
// allow the newest stage (6 async ops/thread) to remain in flight
static __device__ __forceinline__ void wait_async_older() {
  asm volatile("s_wait_asynccnt 6" ::: "memory");
}

// LDS layout per buffer (u16 elements)
#define OFF_AS 0         // A tile: 128 rows x 32 k
#define OFF_BS 4096      // B tile: 256 n-rows (4 gates x 64 cols) x 32 k
#define BUF_ELEMS 12288  // 24 KB per buffer
#define NBUF 3           // 3-deep async pipeline (72 KB LDS)

// ---------------- fused LSTM cell: concatenated-K GEMM (WMMA bf16) + gates ----------------
__global__ __launch_bounds__(256)
void lstm_wmma(const unsigned short* __restrict__ xb,   // [B][IN]  bf16
               const unsigned short* __restrict__ hb,   // [B][H]   bf16
               const unsigned short* __restrict__ wiT,  // [4H][IN] bf16 (transposed)
               const unsigned short* __restrict__ whT,  // [4H][H]  bf16 (transposed)
               const float* __restrict__ c_prev,        // [B][H]
               const float* __restrict__ b_i,           // [4H]
               const float* __restrict__ b_h,           // [4H]
               float* __restrict__ h_out0,
               float* __restrict__ h_out1,
               float* __restrict__ c_out) {
  __shared__ __align__(32) unsigned short smem[NBUF][BUF_ELEMS];

  const int tid    = threadIdx.x;
  const int wave   = tid >> 5;
  const int lane   = tid & 31;
  const int laneLo = lane & 15;
  const int laneHi = lane >> 4;
  const int mg = wave >> 2;          // 0..1 : 64-row half of the 128-row block tile
  const int nt = wave & 3;           // 0..3 : 16-col tile

  const int rowBase = blockIdx.y * 128;  // batch rows
  const int colBase = blockIdx.x * 64;   // hidden cols

  // raw LDS byte offset of smem (hardware adds LDS_BASE for async ops)
  const unsigned lds_base =
      (unsigned)(unsigned long long)(__attribute__((address_space(3))) unsigned short*)smem;

  v8f zero = {0.f, 0.f, 0.f, 0.f, 0.f, 0.f, 0.f, 0.f};
  v8f acc[4][4];                      // [m-subtile][gate]
  #pragma unroll
  for (int mt = 0; mt < 4; ++mt)
    #pragma unroll
    for (int g = 0; g < 4; ++g) acc[mt][g] = zero;

  // A staging: 128 rows x 32 k = 8 KB, 32B (2x b128) per thread.
  const int a_m    = tid >> 1;        // 0..127
  const int a_half = tid & 1;         // k half: 0 -> k 0..15, 1 -> k 16..31
  // B staging: 256 n-rows x 32 k = 16 KB, 64B (4x b128) per thread.
  const int b_g = tid >> 6, b_j = tid & 63;
  const size_t b_row = (size_t)(b_g * DH + colBase + b_j);

  // stage tile t (6 async b128 per thread) into buffer b
  auto stage = [&](int b, int t) {
    const int kk = (t & (NT - 1)) * 32;       // wraps harmlessly on the last 2 iterations
    const unsigned short* Asrc = (kk < KD) ? xb  : hb;
    const unsigned short* Bsrc = (kk < KD) ? wiT : whT;
    const int kc = kk & (KD - 1);
    const unsigned base = lds_base + (unsigned)b * (BUF_ELEMS * 2);
    #pragma unroll
    for (int c = 0; c < 2; ++c) {
      const int k8  = a_half * 16 + c * 8;                    // 0,8,16,24
      const int pos = k8;                                      // contiguous k layout (matches B)
      async_ld_b128(base + (OFF_AS + a_m * 32 + pos) * 2,
                    Asrc + (size_t)(rowBase + a_m) * KD + kc + k8);
    }
    #pragma unroll
    for (int c = 0; c < 4; ++c)
      async_ld_b128(base + (OFF_BS + tid * 32 + c * 8) * 2,
                    Bsrc + b_row * KD + kc + c * 8);
  };

  stage(0, 0);
  stage(1, 1);

  for (int t = 0; t < NT; ++t) {
    wait_async_older();   // everything older than the newest stage done => tile t landed
    __syncthreads();      // all waves: tile t visible; prev-prev buffer fully consumed

    stage((t + 2) % NBUF, t + 2);   // DMA two tiles ahead (overwrites buffer read at t-1)

    const unsigned short* S = smem[t % NBUF];
    // A fragments: 4 M-subtiles, each one contiguous 32B chunk per lane
    v16bf a[4];
    #pragma unroll
    for (int mt = 0; mt < 4; ++mt)
      a[mt] = *(const v16bf*)(S + OFF_AS + (mg * 64 + mt * 16 + laneLo) * 32 + laneHi * 16);
    #pragma unroll
    for (int g = 0; g < 4; ++g) {
      v16bf bb = *(const v16bf*)(S + OFF_BS + (g * 64 + nt * 16 + laneLo) * 32 + laneHi * 16);
      #pragma unroll
      for (int mt = 0; mt < 4; ++mt)
        acc[mt][g] = __builtin_amdgcn_wmma_f32_16x16x32_bf16(
            false, a[mt], false, bb, (short)0, acc[mt][g], false, false);
    }
  }

  // ---- fused epilogue: bias + activations + cell/hidden update (branch-free) ----
  const int j = colBase + nt * 16 + laneLo;   // hidden column
  float bsum[4];
  #pragma unroll
  for (int g = 0; g < 4; ++g) bsum[g] = b_i[g * DH + j] + b_h[g * DH + j];

  #pragma unroll
  for (int mt = 0; mt < 4; ++mt) {
    #pragma unroll
    for (int r = 0; r < 8; ++r) {
      // C/D layout: VGPR r -> row r (lanes 0-15) / row r+8 (lanes 16-31)
      int row = rowBase + mg * 64 + mt * 16 + laneHi * 8 + r;
      float f  = fsigmoid(acc[mt][0][r] + bsum[0]);
      float i_ = fsigmoid(acc[mt][1][r] + bsum[1]);
      float g_ = ftanh   (acc[mt][2][r] + bsum[2]);
      float o  = fsigmoid(acc[mt][3][r] + bsum[3]);
      size_t idx = (size_t)row * DH + j;
      float c = f * c_prev[idx] + i_ * g_;
      float h = o * ftanh(c);
      h_out0[idx] = h;
      h_out1[idx] = h;
      c_out[idx]  = c;
    }
  }
}

extern "C" void kernel_launch(void* const* d_in, const int* in_sizes, int n_in,
                              void* d_out, int out_size, void* d_ws, size_t ws_size,
                              hipStream_t stream) {
  const float* x  = (const float*)d_in[0];
  const float* h1 = (const float*)d_in[1];
  const float* c1 = (const float*)d_in[2];
  const float* Wi = (const float*)d_in[3];
  const float* bi = (const float*)d_in[4];
  const float* Wh = (const float*)d_in[5];
  const float* bh = (const float*)d_in[6];

  unsigned short* xb  = (unsigned short*)d_ws;
  unsigned short* hb  = xb  + (size_t)DB * DI;
  unsigned short* wiT = hb  + (size_t)DB * DH;
  unsigned short* whT = wiT + (size_t)DI * H4;

  float* out  = (float*)d_out;
  float* h0o  = out;
  float* h1o  = out + (size_t)DB * DH;
  float* co   = out + 2 * (size_t)DB * DH;

  cvt_bf16<<<(DB * DI) / 1024, 256, 0, stream>>>(x,  xb);
  cvt_bf16<<<(DB * DH) / 1024, 256, 0, stream>>>(h1, hb);
  dim3 tb(32, 8);
  cvt_tr_bf16<<<dim3(H4 / 32, DI / 32), tb, 0, stream>>>(Wi, wiT, DI, H4);
  cvt_tr_bf16<<<dim3(H4 / 32, DH / 32), tb, 0, stream>>>(Wh, whT, DH, H4);

  lstm_wmma<<<dim3(DH / 64, DB / 128), 256, 0, stream>>>(
      xb, hb, wiT, whT, c1, bi, bh, h0o, h1o, co);
}
